// NeurJudge_12850542149903
// MI455X (gfx1250) — matrix operand order repaired and checked by reading, not compiled, not run on hardware
//
#include <hip/hip_runtime.h>
#include <hip/hip_bf16.h>
#include <math.h>

typedef __attribute__((ext_vector_type(16))) _Float16 v16h;
typedef __attribute__((ext_vector_type(8)))  float    v8f;

#define EPSF 1e-10f

// permutation of k within a 32-wide K block so that a lane's A fragment is a
// contiguous 16-half (32B) LDS read:  p(k) = khi*16 + g*8 + o  where
// k = g*16 + khi*8 + o   (see CDNA5 16-bit A-matrix 16x32 layout)
__device__ __forceinline__ int kperm(int k)
{
    return ((k >> 3) & 1) * 16 + ((k >> 4) & 1) * 8 + (k & 7);
}

// ---------------------------------------------------------------------------
// WMMA GEMM: C[M,N] = A[M,K] * Bmat (+bias), Bmat(k,n) = TRANS ? W[n*ldb+k]
//                                                            : W[k*ldb+n]
// Batched over blockIdx.z with element strides asb/bsb/csb.
// Block = 256 threads = 8 waves; block tile 32x64, wave tile 16x16, K step 32.
// Double-buffered LDS staging (one barrier per K step); staging issues all
// global loads first (register array), then converts/stores, so the loads
// form one clause and overlap the current WMMA.
// REQUIRES: M a multiple of 32 (all call sites satisfy this).
// ---------------------------------------------------------------------------
template<bool TRANS, bool BIAS>
__global__ __launch_bounds__(256) void wmma_gemm(
    const float* __restrict__ A, int lda, long long asb,
    const float* __restrict__ W, int ldb, long long bsb,
    float* __restrict__ C, int ldc, long long csb,
    const float* __restrict__ bias,
    int M, int N, int K)
{
    __shared__ _Float16 As[2][32 * 40];   // [row][k], stride 40 halves
    __shared__ _Float16 Bf[2][64 * 40];   // [n][k],        stride 40 halves

    const float* Ab = A + (size_t)blockIdx.z * (size_t)asb;
    const float* Wb = W + (size_t)blockIdx.z * (size_t)bsb;
    float*       Cb = C + (size_t)blockIdx.z * (size_t)csb;

    int m0 = blockIdx.x * 32, n0 = blockIdx.y * 64;
    int tid  = threadIdx.x;
    int wave = tid >> 5, lane = tid & 31;
    int mt = wave >> 2, nt = wave & 3;       // 2x4 wave grid
    int khi = lane >> 4, l16 = lane & 15;
    int K1 = K - 1, N1 = N - 1;

    auto stage = [&](int buf, int k0) {
        float av[4], bv[8];
        // phase 1: issue ALL global loads (one clause, many outstanding)
#pragma unroll
        for (int it = 0; it < 4; it++) {
            int i = tid + it * 256;           // A tile 32x32
            int r = i >> 5, c = i & 31;
            int k = k0 + c;
            av[it] = Ab[(size_t)(m0 + r) * lda + (k < K ? k : K1)];
        }
#pragma unroll
        for (int it = 0; it < 8; it++) {
            int i = tid + it * 256;           // B tile 32x64
            if (TRANS) {
                int n = i >> 5, c = i & 31;
                int gn = n0 + n, k = k0 + c;
                bv[it] = Wb[(size_t)(gn < N ? gn : N1) * ldb + (k < K ? k : K1)];
            } else {
                int kk = i >> 6, nn = i & 63;
                int k = k0 + kk, gn = n0 + nn;
                bv[it] = Wb[(size_t)(k < K ? k : K1) * ldb + (gn < N ? gn : N1)];
            }
        }
        // phase 2: convert + store to LDS (drains loads progressively)
#pragma unroll
        for (int it = 0; it < 4; it++) {
            int i = tid + it * 256;
            int r = i >> 5, c = i & 31;
            int k = k0 + c;
            As[buf][r * 40 + c] = (k < K) ? (_Float16)av[it] : (_Float16)0.f;
        }
#pragma unroll
        for (int it = 0; it < 8; it++) {
            int i = tid + it * 256;
            if (TRANS) {
                int n = i >> 5, c = i & 31;
                int gn = n0 + n, k = k0 + c;
                Bf[buf][n * 40 + c] = (gn < N && k < K) ? (_Float16)bv[it] : (_Float16)0.f;
            } else {
                int kk = i >> 6, nn = i & 63;
                int k = k0 + kk, gn = n0 + nn;
                Bf[buf][nn * 40 + kk] = (gn < N && k < K) ? (_Float16)bv[it] : (_Float16)0.f;
            }
        }
    };

    v8f acc = {};
    int parity = 0;
    stage(0, 0);
    __syncthreads();
    for (int k0 = 0; k0 < K; k0 += 32) {
        if (k0 + 32 < K) stage(parity ^ 1, k0 + 32);
        v16h a = *(const v16h*)&As[parity][(mt * 16 + l16) * 40 + khi * 16];
        v16h b = *(const v16h*)&Bf[parity][(nt * 16 + l16) * 40 + khi * 16];
        acc = __builtin_amdgcn_wmma_f32_16x16x32_f16(false, a, false, b,
                                                     (short)0, acc, false, false);
        __syncthreads();
        parity ^= 1;
    }
#pragma unroll
    for (int r = 0; r < 8; r++) {
        int m = m0 + mt * 16 + khi * 8 + r;
        int n = n0 + nt * 16 + l16;
        if (n < N) {
            float v = acc[r];
            if (BIAS) v += bias[n];
            Cb[(size_t)m * ldc + n] = v;
        }
    }
}

// ---------------------------------------------------------------------------
// Persistent GRU scan.  One block = one chunk of <=16 sequences, one
// direction.  Per step: phase1 waves compute gh = h @ Whh^T with WMMA
// (Whh pre-converted f16, padded [N16 x K16]); phase2 computes gates and
// updates h.  State kept twice: f32 (recurrence precision) + unpermuted
// f16 copy used as the WMMA A operand (contiguous v16h LDS loads).
// ---------------------------------------------------------------------------
template<int N16, int K16>
__global__ __launch_bounds__(1024) void gru_scan(
    const float* __restrict__ xs,       // [Nseq, T, 3H]
    const _Float16* __restrict__ whh,   // [N16, K16] padded f16
    const float* __restrict__ bhh,      // [3H]
    float* __restrict__ y,              // [Nseq, T, ldy]
    int Nseq, int T, int H, int reverse, int colOff, int ldy)
{
    __shared__ _Float16 gh16[16 * N16];        // gate pre-activations (f16)
    __shared__ float    hb32[16 * K16];        // f32 recurrent state
    __shared__ _Float16 hb16[16 * (K16 + 8)];  // f16 state [m][k] (A operand)

    int tid = threadIdx.x;
    int s0  = blockIdx.x * 16;
    int Mc  = Nseq - s0; if (Mc > 16) Mc = 16;
    int N3  = 3 * H;
    int ntiles = (N3 + 15) >> 4;
    int ksteps = (H + 31) >> 5;
    const int HS = K16 + 8;                    // hb16 row stride (halves)

    for (int i = tid; i < 16 * K16; i += 1024) hb32[i] = 0.f;
    for (int i = tid; i < 16 * HS;  i += 1024) hb16[i] = (_Float16)0.f;
    __syncthreads();

    int wave = tid >> 5, lane = tid & 31;
    int khi = lane >> 4, l16 = lane & 15;

    for (int s = 0; s < T; s++) {
        int t = reverse ? (T - 1 - s) : s;
        // phase 1: gh tiles via WMMA
        for (int tile = wave; tile < ntiles; tile += 32) {
            v8f acc = {};
            const _Float16* wrow = whh + (size_t)(tile * 16 + l16) * K16 + khi * 16;
            const _Float16* arow = hb16 + l16 * HS + khi * 16;
            for (int ks = 0; ks < ksteps; ks++) {
                v16h a = *(const v16h*)(arow + ks * 32);
                v16h b = *(const v16h*)(wrow + ks * 32);
                acc = __builtin_amdgcn_wmma_f32_16x16x32_f16(false, a, false, b,
                                                             (short)0, acc, false, false);
            }
#pragma unroll
            for (int r = 0; r < 8; r++)
                gh16[(khi * 8 + r) * N16 + tile * 16 + l16] = (_Float16)acc[r];
        }
        __syncthreads();
        // phase 2: gates
        int tot = Mc * H;
        for (int e = tid; e < tot; e += 1024) {
            int m = e / H, j = e - m * H;
            size_t row = (size_t)(s0 + m) * T + t;
            const float* xr = xs + row * N3;
            float gr = (float)gh16[m * N16 + j]         + bhh[j];
            float gz = (float)gh16[m * N16 + H + j]     + bhh[H + j];
            float gn = (float)gh16[m * N16 + 2 * H + j] + bhh[2 * H + j];
            float r_ = 1.f / (1.f + expf(-(xr[j] + gr)));
            float z_ = 1.f / (1.f + expf(-(xr[H + j] + gz)));
            float n_ = tanhf(xr[2 * H + j] + r_ * gn);
            float hp = hb32[m * K16 + j];
            float hn = (1.f - z_) * n_ + z_ * hp;
            hb32[m * K16 + j] = hn;
            hb16[m * HS + j] = (_Float16)hn;
            y[row * ldy + colOff + j] = hn;
        }
        __syncthreads();
    }
}

// ---------------------------------------------------------------------------
// small kernels
// ---------------------------------------------------------------------------
__global__ void conv_f16_pad(const float* __restrict__ W, int N, int K,
                             _Float16* __restrict__ out, int N16, int K16)
{
    size_t i = (size_t)blockIdx.x * blockDim.x + threadIdx.x;
    size_t tot = (size_t)N16 * K16;
    if (i >= tot) return;
    int n = (int)(i / K16), k = (int)(i % K16);
    out[i] = (n < N && k < K) ? (_Float16)W[(size_t)n * K + k] : (_Float16)0.f;
}

__global__ void gather_embed(const float* __restrict__ emb,
                             const int* __restrict__ tok,
                             float* __restrict__ out, int rows, int D)
{
    int r = blockIdx.x;
    if (r >= rows) return;
    const float* src = emb + (size_t)tok[r] * D;
    for (int d = threadIdx.x; d < D; d += blockDim.x)
        out[(size_t)r * D + d] = src[d];
}

__global__ void gather_sel_tokens(const int* __restrict__ sel,
                                  const int* __restrict__ toks, int L,
                                  int* __restrict__ out)
{
    int b = blockIdx.x;
    for (int l = threadIdx.x; l < L; l += blockDim.x)
        out[b * L + l] = toks[(size_t)sel[b] * L + l];
}

__global__ void mean_rows(const float* __restrict__ in, int ld,
                          float* __restrict__ out, int N, int T, int D)
{
    int n = blockIdx.x;
    if (n >= N) return;
    for (int d = threadIdx.x; d < D; d += blockDim.x) {
        float s = 0.f;
        for (int t = 0; t < T; t++) s += in[((size_t)n * T + t) * ld + d];
        out[(size_t)n * D + d] = s / (float)T;
    }
}

__global__ void copy_cols(const float* __restrict__ src, int lds_, int sOff,
                          float* __restrict__ dst, int ldd, int dOff,
                          int rows, int cols)
{
    size_t i = (size_t)blockIdx.x * blockDim.x + threadIdx.x;
    size_t tot = (size_t)rows * cols;
    if (i >= tot) return;
    size_t r = i / cols; int c = (int)(i % cols);
    dst[r * ldd + dOff + c] = src[r * lds_ + sOff + c];
}

__global__ void tile_cols(const float* __restrict__ src, int Cs,
                          float* __restrict__ dst, int ldd, int dOff,
                          int Bn, int Tn, int cols)
{
    size_t i = (size_t)blockIdx.x * blockDim.x + threadIdx.x;
    size_t tot = (size_t)Bn * Tn * cols;
    if (i >= tot) return;
    int c = (int)(i % cols); size_t r = i / cols; int b = (int)(r / Tn);
    dst[r * ldd + dOff + c] = src[(size_t)b * Cs + c];
}

__global__ void graph_layer(const float* __restrict__ labIn,
                            float* __restrict__ labOut,
                            const int* __restrict__ nb,
                            const int* __restrict__ mask, int Ncnt, int Kn)
{
    int i = blockIdx.x;
    if (i >= Ncnt) return;
    const int Dd = 300;
    __shared__ float red1[256], red2[256];
    __shared__ float coef[16];
    __shared__ float degS;
    int tid = threadIdx.x;
    for (int j = 0; j < Kn; j++) {
        int nbj = nb[i * Kn + j];
        float s1 = 0.f, s2 = 0.f;
        for (int d = tid; d < Dd; d += 256) {
            float lj = labIn[(size_t)nbj * Dd + d];
            s1 += labIn[(size_t)i * Dd + d] * lj;
            s2 += lj * lj;
        }
        red1[tid] = s1; red2[tid] = s2; __syncthreads();
        for (int st = 128; st > 0; st >>= 1) {
            if (tid < st) { red1[tid] += red1[tid + st]; red2[tid] += red2[tid + st]; }
            __syncthreads();
        }
        if (tid == 0) coef[j] = red1[0] / (red2[0] + EPSF);
        __syncthreads();
    }
    if (tid == 0) {
        float dg = 0.f;
        for (int j = 0; j < Kn; j++) dg += (float)mask[i * Kn + j];
        degS = dg;
    }
    __syncthreads();
    float deg = degS;
    float inv = 1.f / fmaxf(deg, 1.f);
    for (int d = tid; d < Dd; d += 256) {
        float mean = 0.f;
        for (int j = 0; j < Kn; j++) {
            float m = (float)mask[i * Kn + j];
            mean += m * coef[j] * labIn[(size_t)nb[i * Kn + j] * Dd + d];
        }
        mean *= inv;
        float v = labIn[(size_t)i * Dd + d];
        labOut[(size_t)i * Dd + d] = (deg > 0.f) ? (v - mean) : v;
    }
}

__global__ void rowmax(const float* __restrict__ S, float* __restrict__ mx,
                       int rows, int C)
{
    int r = blockIdx.x * blockDim.x + threadIdx.x;
    if (r >= rows) return;
    const float* p = S + (size_t)r * C;
    float m = -INFINITY;
    for (int c = 0; c < C; c++) m = fmaxf(m, p[c]);
    mx[r] = m;
}

__global__ void softmax_T(const float* __restrict__ mx, float* __restrict__ att,
                          int T)
{
    int b = blockIdx.x;
    int t = threadIdx.x;                 // blockDim.x == T (512)
    __shared__ float red[512];
    float v = mx[(size_t)b * T + t];
    red[t] = v; __syncthreads();
    for (int st = 256; st > 0; st >>= 1) {
        if (t < st) red[t] = fmaxf(red[t], red[t + st]);
        __syncthreads();
    }
    float m = red[0]; __syncthreads();
    float e = expf(v - m);
    red[t] = e; __syncthreads();
    for (int st = 256; st > 0; st >>= 1) {
        if (t < st) red[t] += red[t + st];
        __syncthreads();
    }
    att[(size_t)b * T + t] = e / red[0];
}

__global__ void wsum(const float* __restrict__ att, const float* __restrict__ dh,
                     float* __restrict__ out, int T, int D)
{
    int b = blockIdx.x;
    for (int d = threadIdx.x; d < D; d += blockDim.x) {
        float s = 0.f;
        for (int t = 0; t < T; t++)
            s += att[(size_t)b * T + t] * dh[((size_t)b * T + t) * D + d];
        out[(size_t)b * D + d] = s;
    }
}

__global__ void argmax_row(const float* __restrict__ X, int C,
                           int* __restrict__ out)
{
    int r = blockIdx.x;
    __shared__ float bv[128];
    __shared__ int   bi[128];
    int tid = threadIdx.x;
    float best = -INFINITY; int bidx = 0;
    for (int c = tid; c < C; c += 128) {
        float v = X[(size_t)r * C + c];
        if (v > best) { best = v; bidx = c; }
    }
    bv[tid] = best; bi[tid] = bidx; __syncthreads();
    for (int st = 64; st > 0; st >>= 1) {
        if (tid < st) {
            if (bv[tid + st] > bv[tid] ||
                (bv[tid + st] == bv[tid] && bi[tid + st] < bi[tid])) {
                bv[tid] = bv[tid + st]; bi[tid] = bi[tid + st];
            }
        }
        __syncthreads();
    }
    if (tid == 0) out[r] = bi[0];
}

__global__ void mask_softmax(float* __restrict__ S, int rows, int L)
{
    int r = blockIdx.x * blockDim.x + threadIdx.x;
    if (r >= rows) return;
    float* row = S + (size_t)r * L;
    float mx = -INFINITY;
    for (int l = 0; l < L; l++) { float v = row[l]; if (v != 0.f && v > mx) mx = v; }
    if (mx == -INFINITY) { for (int l = 0; l < L; l++) row[l] = 0.f; return; }
    float sum = 0.f;
    for (int l = 0; l < L; l++) {
        float v = row[l];
        float e = (v == 0.f) ? 0.f : expf(v - mx);
        row[l] = e; sum += e;
    }
    float inv = 1.f / sum;
    for (int l = 0; l < L; l++) row[l] *= inv;
}

__global__ void fact_sep(const float* __restrict__ scen,
                         const float* __restrict__ circ, int ldcir,
                         float* __restrict__ simDst, int ldS, int offS,
                         float* __restrict__ difDst, int ldD, int offD)
{
    int r = blockIdx.x;
    const int Dd = 300;
    __shared__ float r1[128], r2[128];
    int tid = threadIdx.x;
    float s1 = 0.f, s2 = 0.f;
    for (int d = tid; d < Dd; d += 128) {
        float sc = scen[(size_t)r * Dd + d];
        s1 += circ[(size_t)r * ldcir + d] * sc;
        s2 += sc * sc;
    }
    r1[tid] = s1; r2[tid] = s2; __syncthreads();
    for (int st = 64; st > 0; st >>= 1) {
        if (tid < st) { r1[tid] += r1[tid + st]; r2[tid] += r2[tid + st]; }
        __syncthreads();
    }
    float c = r1[0] / (r2[0] + EPSF);
    for (int d = tid; d < Dd; d += 128) {
        float sim = c * scen[(size_t)r * Dd + d];
        simDst[(size_t)r * ldS + offS + d] = sim;
        difDst[(size_t)r * ldD + offD + d] = circ[(size_t)r * ldcir + d] - sim;
    }
}

// ---------------------------------------------------------------------------
// host side
// ---------------------------------------------------------------------------
extern "C" void kernel_launch(void* const* d_in, const int* in_sizes, int n_in,
                              void* d_out, int out_size, void* d_ws, size_t ws_size,
                              hipStream_t stream)
{
    (void)in_sizes; (void)n_in; (void)out_size; (void)ws_size;
    constexpr int V = 339503, D = 200, Hh = 150, B = 32, T = 512;
    constexpr int Nc = 119, Na = 103, Kn = 8, Lc = 32, La = 64, Lv = 50;
    constexpr int Dh = 300;                       // 2*Hh
    constexpr int Ht = 450, Ha = 600;             // term/art hidden
    (void)V;

    const float* emb       = (const float*)d_in[0];
    const float* enc_Wih   = (const float*)d_in[1];
    const float* enc_Whh   = (const float*)d_in[2];
    const float* enc_bih   = (const float*)d_in[3];
    const float* enc_bhh   = (const float*)d_in[4];
    const float* ech_Wih   = (const float*)d_in[5];
    const float* ech_Whh   = (const float*)d_in[6];
    const float* ech_bih   = (const float*)d_in[7];
    const float* ech_bhh   = (const float*)d_in[8];
    const float* term_Wih  = (const float*)d_in[9];
    const float* term_Whh  = (const float*)d_in[10];
    const float* term_bih  = (const float*)d_in[11];
    const float* term_bhh  = (const float*)d_in[12];
    const float* art_Wih   = (const float*)d_in[13];
    const float* art_Whh   = (const float*)d_in[14];
    const float* art_bih   = (const float*)d_in[15];
    const float* art_bhh   = (const float*)d_in[16];
    const float* Wc        = (const float*)d_in[17];
    const float* bc        = (const float*)d_in[18];
    const float* Wa        = (const float*)d_in[19];
    const float* ba        = (const float*)d_in[20];
    const float* Wt        = (const float*)d_in[21];
    const float* bt        = (const float*)d_in[22];
    const int* documents   = (const int*)d_in[23];
    const int* charge_tok  = (const int*)d_in[24];
    const int* article_tok = (const int*)d_in[25];
    const int* charge_nb   = (const int*)d_in[26];
    const int* charge_nbm  = (const int*)d_in[27];
    const int* article_nb  = (const int*)d_in[28];
    const int* article_nbm = (const int*)d_in[29];
    const int* vct         = (const int*)d_in[30];
    const int* vat         = (const int*)d_in[31];
    float* out             = (float*)d_out;

    // ---- workspace layout ----
    char* ws = (char*)d_ws;
    size_t off = 0;
    auto alloc = [&](size_t bytes) {
        size_t o = off; off += (bytes + 255) & ~(size_t)255; return o;
    };
    const int N16e = 464, K16e = 160;   // H=150 (3H=450)
    const int N16t = 1360, K16t = 480;  // H=450 (3H=1350)
    const int N16a = 1808, K16a = 608;  // H=600 (3H=1800)
    size_t oWE = alloc(2ull * N16e * K16e * 2);   // enc
    size_t oWC = alloc(2ull * N16e * K16e * 2);   // encch
    size_t oWT = alloc(2ull * N16t * K16t * 2);   // term
    size_t oWA = alloc(2ull * N16a * K16a * 2);   // art
    size_t oDH   = alloc(4ull * B * T * Dh);
    size_t oSEC  = alloc(4ull * B * T * Dh);
    size_t oGATH = alloc(4ull * B * T * D);
    size_t oENCH = alloc(4ull * Na * La * Dh);    // >= Nc*Lc*Dh
    size_t oORIA = alloc(4ull * Nc * Dh);
    size_t oLABA = alloc(4ull * Nc * Dh);
    size_t oLABB = alloc(4ull * Nc * Dh);
    size_t oORIB = alloc(4ull * Na * Dh);
    size_t oLABC = alloc(4ull * Na * Dh);
    size_t oLABD = alloc(4ull * Na * Dh);
    size_t oDMEAN = alloc(4ull * B * Dh);
    size_t oDHC   = alloc(4ull * B * Dh);
    size_t oDA    = alloc(4ull * B * Dh);
    size_t oDHA   = alloc(4ull * B * Dh);
    size_t oDB    = alloc(4ull * B * Dh);
    size_t oFEAT  = alloc(4ull * B * 1200);
    size_t oMX    = alloc(4ull * B * T);
    size_t oATT   = alloc(4ull * B * T);
    size_t oCP    = alloc(4ull * B);
    size_t oAP    = alloc(4ull * B);
    size_t oVTOK  = alloc(4ull * B * Lv);
    size_t oVCH   = alloc(4ull * B * Lv * Dh);
    size_t oBIG1  = alloc(4ull * B * T * 1800);   // xs / scores
    size_t oBIG2  = alloc(4ull * B * T * 1200);   // y / scenario
    size_t oBIG3  = alloc(4ull * B * T * 1200);   // concat inputs

    auto F = [&](size_t o) { return (float*)(ws + o); };
    auto H16 = [&](size_t o) { return (_Float16*)(ws + o); };
    auto I = [&](size_t o) { return (int*)(ws + o); };

    // ---- launch helpers ----
    auto gemm = [&](const float* A, int lda, long long asb,
                    const float* W, int ldb, long long bsb,
                    float* C, int ldc, long long csb, const float* bias,
                    int M, int N, int K, bool trans, int batch) {
        dim3 g((M + 31) / 32, (N + 63) / 64, batch), blk(256);
        if (trans) {
            if (bias) wmma_gemm<true, true><<<g, blk, 0, stream>>>(A, lda, asb, W, ldb, bsb, C, ldc, csb, bias, M, N, K);
            else      wmma_gemm<true, false><<<g, blk, 0, stream>>>(A, lda, asb, W, ldb, bsb, C, ldc, csb, bias, M, N, K);
        } else {
            wmma_gemm<false, false><<<g, blk, 0, stream>>>(A, lda, asb, W, ldb, bsb, C, ldc, csb, bias, M, N, K);
        }
    };
    auto scan = [&](const float* xs, const _Float16* whh, const float* bhh,
                    float* y, int Nseq, int Tt, int H, int rev, int colOff, int ldy) {
        int chunks = (Nseq + 15) / 16;
        if (H == 150)      gru_scan<464, 160><<<chunks, 1024, 0, stream>>>(xs, whh, bhh, y, Nseq, Tt, H, rev, colOff, ldy);
        else if (H == 450) gru_scan<1360, 480><<<chunks, 1024, 0, stream>>>(xs, whh, bhh, y, Nseq, Tt, H, rev, colOff, ldy);
        else               gru_scan<1808, 608><<<chunks, 1024, 0, stream>>>(xs, whh, bhh, y, Nseq, Tt, H, rev, colOff, ldy);
    };
    auto bigru = [&](const float* x, int Nseq, int Tt, int Din, int H,
                     const float* Wih, const float* bih,
                     const _Float16* whh16, size_t whhStride, const float* bhh,
                     float* xsbuf, float* y) {
        int N3 = 3 * H;
        for (int dir = 0; dir < 2; dir++) {
            gemm(x, Din, 0, Wih + (size_t)dir * N3 * Din, Din, 0,
                 xsbuf, N3, 0, bih + (size_t)dir * N3,
                 Nseq * Tt, N3, Din, true, 1);
            scan(xsbuf, whh16 + (size_t)dir * whhStride, bhh + (size_t)dir * N3,
                 y, Nseq, Tt, H, dir, dir * H, 2 * H);
        }
    };
    auto codewise = [&](const float* labels, int Nl, float* outBD) {
        gemm(F(oDH), Dh, 0, labels, Dh, 0, F(oBIG1), Nl, 0, nullptr,
             B * T, Nl, Dh, true, 1);
        rowmax<<<(B * T + 255) / 256, 256, 0, stream>>>(F(oBIG1), F(oMX), B * T, Nl);
        softmax_T<<<B, 512, 0, stream>>>(F(oMX), F(oATT), T);
        wsum<<<B, 256, 0, stream>>>(F(oATT), F(oDH), outBD, T, Dh);
    };
    auto factsep = [&](const float* vh, const float* circ,
                       float* simDst, int ldS, int offS,
                       float* difDst, int ldD, int offD) {
        // batched scores: [b] A=circ_b [T,300], W=vh_b [Lv,300] (TN)
        gemm(circ, Dh, (long long)T * Dh, vh, Dh, (long long)Lv * Dh,
             F(oBIG1), Lv, (long long)T * Lv, nullptr, T, Lv, Dh, true, B);
        mask_softmax<<<(B * T + 255) / 256, 256, 0, stream>>>(F(oBIG1), B * T, Lv);
        // scenario = sm @ vh (NN), batched
        gemm(F(oBIG1), Lv, (long long)T * Lv, vh, Dh, (long long)Lv * Dh,
             F(oBIG2), Dh, (long long)T * Dh, nullptr, T, Dh, Lv, false, B);
        fact_sep<<<B * T, 128, 0, stream>>>(F(oBIG2), circ, Dh,
                                            simDst, ldS, offS, difDst, ldD, offD);
    };
    auto conv = [&](const float* W, int N, int K, _Float16* outw, int N16, int K16) {
        size_t tot = (size_t)N16 * K16;
        conv_f16_pad<<<(unsigned)((tot + 255) / 256), 256, 0, stream>>>(W, N, K, outw, N16, K16);
    };

    // ---- pre-convert Whh to padded f16 ----
    for (int d = 0; d < 2; d++) {
        conv(enc_Whh  + (size_t)d * 450 * 150,  450, 150, H16(oWE) + (size_t)d * N16e * K16e, N16e, K16e);
        conv(ech_Whh  + (size_t)d * 450 * 150,  450, 150, H16(oWC) + (size_t)d * N16e * K16e, N16e, K16e);
        conv(term_Whh + (size_t)d * 1350 * 450, 1350, 450, H16(oWT) + (size_t)d * N16t * K16t, N16t, K16t);
        conv(art_Whh  + (size_t)d * 1800 * 600, 1800, 600, H16(oWA) + (size_t)d * N16a * K16a, N16a, K16a);
    }

    // ---- charge / article token encoders (encch params) ----
    gather_embed<<<Nc * Lc, 256, 0, stream>>>(emb, charge_tok, F(oGATH), Nc * Lc, D);
    bigru(F(oGATH), Nc, Lc, D, Hh, ech_Wih, ech_bih, H16(oWC), (size_t)N16e * K16e,
          ech_bhh, F(oBIG1), F(oENCH));
    mean_rows<<<Nc, 256, 0, stream>>>(F(oENCH), Dh, F(oORIA), Nc, Lc, Dh);

    gather_embed<<<Na * La, 256, 0, stream>>>(emb, article_tok, F(oGATH), Na * La, D);
    bigru(F(oGATH), Na, La, D, Hh, ech_Wih, ech_bih, H16(oWC), (size_t)N16e * K16e,
          ech_bhh, F(oBIG1), F(oENCH));
    mean_rows<<<Na, 256, 0, stream>>>(F(oENCH), Dh, F(oORIB), Na, La, Dh);

    // ---- graph decomposition (2 layers each) ----
    copy_cols<<<(Nc * Dh + 255) / 256, 256, 0, stream>>>(F(oORIA), Dh, 0, F(oLABA), Dh, 0, Nc, Dh);
    graph_layer<<<Nc, 256, 0, stream>>>(F(oLABA), F(oLABB), charge_nb, charge_nbm, Nc, Kn);
    graph_layer<<<Nc, 256, 0, stream>>>(F(oLABB), F(oLABA), charge_nb, charge_nbm, Nc, Kn);
    copy_cols<<<(Na * Dh + 255) / 256, 256, 0, stream>>>(F(oORIB), Dh, 0, F(oLABC), Dh, 0, Na, Dh);
    graph_layer<<<Na, 256, 0, stream>>>(F(oLABC), F(oLABD), article_nb, article_nbm, Na, Kn);
    graph_layer<<<Na, 256, 0, stream>>>(F(oLABD), F(oLABC), article_nb, article_nbm, Na, Kn);

    // ---- document encoder (enc params) ----
    gather_embed<<<B * T, 256, 0, stream>>>(emb, documents, F(oGATH), B * T, D);
    bigru(F(oGATH), B, T, D, Hh, enc_Wih, enc_bih, H16(oWE), (size_t)N16e * K16e,
          enc_bhh, F(oBIG1), F(oDH));
    mean_rows<<<B, 256, 0, stream>>>(F(oDH), Dh, F(oDMEAN), B, T, Dh);

    // ---- code-wise attention + charge head ----
    codewise(F(oLABA), Nc, F(oDHC));   // new_charge
    codewise(F(oORIA), Nc, F(oDA));    // ori_a
    copy_cols<<<(B * Dh + 255) / 256, 256, 0, stream>>>(F(oDMEAN), Dh, 0, F(oFEAT), 900, 0, B, Dh);
    copy_cols<<<(B * Dh + 255) / 256, 256, 0, stream>>>(F(oDHC),   Dh, 0, F(oFEAT), 900, 300, B, Dh);
    copy_cols<<<(B * Dh + 255) / 256, 256, 0, stream>>>(F(oDA),    Dh, 0, F(oFEAT), 900, 600, B, Dh);
    gemm(F(oFEAT), 900, 0, Wc, 900, 0, out, Nc, 0, bc, B, Nc, 900, true, 1);
    argmax_row<<<B, 128, 0, stream>>>(out, Nc, I(oCP));

    // ---- verdict charge encoder ----
    gather_sel_tokens<<<B, 64, 0, stream>>>(I(oCP), vct, Lv, I(oVTOK));
    gather_embed<<<B * Lv, 256, 0, stream>>>(emb, I(oVTOK), F(oGATH), B * Lv, D);
    bigru(F(oGATH), B, Lv, D, Hh, enc_Wih, enc_bih, H16(oWE), (size_t)N16e * K16e,
          enc_bhh, F(oBIG1), F(oVCH));

    // ---- fact separation 1: adc -> fact_article col 600; sec ----
    factsep(F(oVCH), F(oDH), F(oBIG3), 1200, 600, F(oSEC), Dh, 0);

    // ---- article branch ----
    codewise(F(oLABC), Na, F(oDHA));   // new_article
    codewise(F(oORIB), Na, F(oDB));    // ori_b
    copy_cols<<<(B * T * Dh + 255) / 256, 256, 0, stream>>>(F(oDH), Dh, 0, F(oBIG3), 1200, 0, B * T, Dh);
    tile_cols<<<(B * T * Dh + 255) / 256, 256, 0, stream>>>(F(oDHA), Dh, F(oBIG3), 1200, 300, B, T, Dh);
    tile_cols<<<(B * T * Dh + 255) / 256, 256, 0, stream>>>(F(oDB),  Dh, F(oBIG3), 1200, 900, B, T, Dh);
    bigru(F(oBIG3), B, T, 1200, Ha, art_Wih, art_bih, H16(oWA), (size_t)N16a * K16a,
          art_bhh, F(oBIG1), F(oBIG2));
    mean_rows<<<B, 256, 0, stream>>>(F(oBIG2), 1200, F(oFEAT), B, T, 1200);
    gemm(F(oFEAT), 1200, 0, Wa, 1200, 0, out + B * Nc, Na, 0, ba, B, Na, 1200, true, 1);
    argmax_row<<<B, 128, 0, stream>>>(out + B * Nc, Na, I(oAP));

    // ---- verdict article encoder ----
    gather_sel_tokens<<<B, 64, 0, stream>>>(I(oAP), vat, Lv, I(oVTOK));
    gather_embed<<<B * Lv, 256, 0, stream>>>(emb, I(oVTOK), F(oGATH), B * Lv, D);
    bigru(F(oGATH), B, Lv, D, Hh, enc_Wih, enc_bih, H16(oWE), (size_t)N16e * K16e,
          enc_bhh, F(oBIG1), F(oVCH));

    // ---- fact separation 2 + term branch ----
    copy_cols<<<(B * T * Dh + 255) / 256, 256, 0, stream>>>(F(oDH), Dh, 0, F(oBIG3), 900, 0, B * T, Dh);
    factsep(F(oVCH), F(oSEC), F(oBIG3), 900, 300, F(oBIG3), 900, 600);
    bigru(F(oBIG3), B, T, 900, Ht, term_Wih, term_bih, H16(oWT), (size_t)N16t * K16t,
          term_bhh, F(oBIG1), F(oBIG2));
    mean_rows<<<B, 256, 0, stream>>>(F(oBIG2), 900, F(oFEAT), B, T, 900);
    gemm(F(oFEAT), 900, 0, Wt, 900, 0, out + B * (Nc + Na), 11, 0, bt, B, 11, 900, true, 1);
}
